// MultiHeadAttention_57939108823734
// MI455X (gfx1250) — compile-verified, matrix operand order checked by reading, not run on hardware
//
#include <hip/hip_runtime.h>
#include <hip/hip_bf16.h>

// ---------------------------------------------------------------------------
// MI455X (gfx1250) fused MHA + LayerNorm, bf16 WMMA pipeline.
// Wave32, v_wmma_f32_16x16x32_bf16 for all matmuls, big-LDS one-pass softmax.
// ---------------------------------------------------------------------------

typedef __bf16 bf16;
typedef __attribute__((ext_vector_type(16))) __bf16 v16bf;
typedef __attribute__((ext_vector_type(8)))  __bf16 v8bf;
typedef __attribute__((ext_vector_type(8)))  float  v8f;

constexpr int BATCH = 2;
constexpr int SEQ   = 2048;
constexpr int DMODEL= 1024;
constexpr int NH    = 16;
constexpr int HD    = 64;     // DK == DV == 64
constexpr int MROWS = BATCH * SEQ;          // 4096

// ---- WMMA helpers ---------------------------------------------------------

__device__ __forceinline__ v8f wmma_bf16(v16bf a, v16bf b, v8f c) {
  // D = A(16x32) * B(32x16) + C(16x16), f32 accumulate
  return __builtin_amdgcn_wmma_f32_16x16x32_bf16(
      /*neg_a=*/false, a, /*neg_b=*/false, b,
      /*c_mod=*/(short)0, c, /*reuse_a=*/false, /*reuse_b=*/false);
}

// Load a 16x32 bf16 A-fragment (or B-fragment from an N-major [16][32+] tile).
// ISA 7.12.2: lanes 0-15 hold rows M=0..15 with K=0..7 / 16..23,
//             lanes 16-31 hold rows M=0..15 with K=8..15 / 24..31.
__device__ __forceinline__ v16bf load_frag(const bf16* p, int stride) {
  const int lane = threadIdx.x & 31;
  const int row  = lane & 15;
  const int kh   = (lane >> 4) << 3;   // 0 or 8
  union { v16bf v; v8bf h[2]; } u;
  u.h[0] = *(const v8bf*)(p + row * stride + kh);        // K = kh .. kh+7
  u.h[1] = *(const v8bf*)(p + row * stride + 16 + kh);   // K = 16+kh .. 16+kh+7
  return u.v;
}

// ---- small conversion kernels --------------------------------------------

__global__ void cvt_f32_bf16(const float* __restrict__ in, bf16* __restrict__ out, int n) {
  int i = (blockIdx.x * blockDim.x + threadIdx.x) * 4;
  if (i < n) {
    float4 v = *(const float4*)(in + i);
    out[i + 0] = (bf16)v.x; out[i + 1] = (bf16)v.y;
    out[i + 2] = (bf16)v.z; out[i + 3] = (bf16)v.w;
  }
}

// [K][N] f32 -> [N][K] bf16  (weights become N-major for WMMA B-fragments)
__global__ void tcvt_f32_bf16(const float* __restrict__ in, bf16* __restrict__ out,
                              int K, int N) {
  int i = blockIdx.x * blockDim.x + threadIdx.x;
  if (i < K * N) {
    int k = i / N, n = i - k * N;
    out[(size_t)n * K + k] = (bf16)in[i];
  }
}

// ---- generic 128x128 bf16 WMMA GEMM --------------------------------------
// A[M][K] row-major bf16, BT[N][K] row-major bf16 (i.e. B transposed).
// MODE 0: out = A*B + bias, scatter bf16 to [B,H,S,64]   (Q/K head split)
// MODE 1: out = A*B + bias, scatter bf16 to [B,H,64,S]   (V head split, transposed)
// MODE 2: out = A*B + bias + resid, f32 row-major [M][N] (output projection)

template <int MODE>
__global__ __launch_bounds__(256) void gemm128(
    const bf16*  __restrict__ A,    const bf16* __restrict__ BT,
    const float* __restrict__ bias, const float* __restrict__ resid,
    float* __restrict__ outF, bf16* __restrict__ outH,
    int M, int N, int K) {
  __shared__ bf16 As[128][40];   // 128x32 tile, stride 40 (bank-friendly, 16B aligned)
  __shared__ bf16 Bs[128][40];

  const int m0 = blockIdx.y * 128;
  const int n0 = blockIdx.x * 128;
  const int tid  = threadIdx.x;
  const int lane = tid & 31;
  const int wave = tid >> 5;        // 8 waves: 2 (M) x 4 (N)
  const int wm = wave >> 2;         // 0..1 -> 64 rows each
  const int wn = wave & 3;          // 0..3 -> 32 cols each

  v8f acc[4][2] = {};

  for (int kt = 0; kt < K; kt += 32) {
    // stage 128x32 of A and of BT (each thread copies 16 bf16 of each)
    const int srow = tid >> 1;
    const int scb  = (tid & 1) * 16;
    {
      const bf16* a = A + (size_t)(m0 + srow) * K + kt + scb;
      *(v8bf*)&As[srow][scb]     = *(const v8bf*)a;
      *(v8bf*)&As[srow][scb + 8] = *(const v8bf*)(a + 8);
      const bf16* b = BT + (size_t)(n0 + srow) * K + kt + scb;
      *(v8bf*)&Bs[srow][scb]     = *(const v8bf*)b;
      *(v8bf*)&Bs[srow][scb + 8] = *(const v8bf*)(b + 8);
    }
    __syncthreads();

    v16bf af[4], bfg[2];
#pragma unroll
    for (int mi = 0; mi < 4; ++mi) af[mi]  = load_frag(&As[wm * 64 + mi * 16][0], 40);
#pragma unroll
    for (int nj = 0; nj < 2; ++nj) bfg[nj] = load_frag(&Bs[wn * 32 + nj * 16][0], 40);
#pragma unroll
    for (int mi = 0; mi < 4; ++mi)
#pragma unroll
      for (int nj = 0; nj < 2; ++nj)
        acc[mi][nj] = wmma_bf16(af[mi], bfg[nj], acc[mi][nj]);
    __syncthreads();
  }

  // epilogue: C layout — VGPR r holds row (r + 8*(lane>=16)), col = lane&15
#pragma unroll
  for (int mi = 0; mi < 4; ++mi) {
#pragma unroll
    for (int nj = 0; nj < 2; ++nj) {
      const int gn = n0 + wn * 32 + nj * 16 + (lane & 15);
#pragma unroll
      for (int r = 0; r < 8; ++r) {
        const int gm = m0 + wm * 64 + mi * 16 + r + ((lane >> 4) << 3);
        float vv = acc[mi][nj][r] + bias[gn];
        if (MODE == 0) {
          const int b = gm >> 11, s = gm & (SEQ - 1);
          const int h = gn >> 6,  d = gn & 63;
          outH[(((size_t)(b * NH + h)) * SEQ + s) * HD + d] = (bf16)vv;
        } else if (MODE == 1) {
          const int b = gm >> 11, s = gm & (SEQ - 1);
          const int h = gn >> 6,  d = gn & 63;
          outH[(((size_t)(b * NH + h)) * HD + d) * SEQ + s] = (bf16)vv;
        } else {
          outF[(size_t)gm * N + gn] = vv + resid[(size_t)gm * N + gn];
        }
      }
    }
  }
}

// ---- scores + mask + softmax, one pass through 131 KB of LDS -------------
// Block: 256 threads (8 waves), one (b,h) and a 16-row Q stripe.
// scores[16][2048] f32 live entirely in LDS; only normalized attn hits HBM.

constexpr int SCPAD = SEQ + 4;   // stride 2052: rows 8 apart land on different banks

__global__ __launch_bounds__(256) void attn_scores_softmax(
    const bf16* __restrict__ qh, const bf16* __restrict__ kh,
    const unsigned char* __restrict__ mask, float* __restrict__ attn) {
  __shared__ float sc[16][SCPAD];          // ~131 KB
  __shared__ bf16  qa[16][72];             // 16x64 Q tile (stride 72, 16B aligned)

  const int bh = blockIdx.y;               // b*NH + h
  const int q0 = blockIdx.x * 16;
  const int b  = bh >> 4;
  const int tid  = threadIdx.x;
  const int lane = tid & 31;
  const int wave = tid >> 5;

  // stage 16x64 Q tile (bf16, already head-split)
  {
    const int row = tid >> 4, cb = (tid & 15) * 4;
    const bf16* src = qh + (((size_t)bh * SEQ) + q0 + row) * HD + cb;
    qa[row][cb + 0] = src[0]; qa[row][cb + 1] = src[1];
    qa[row][cb + 2] = src[2]; qa[row][cb + 3] = src[3];
  }
  __syncthreads();

  const v16bf a0 = load_frag(&qa[0][0],  72);   // K(d) = 0..31
  const v16bf a1 = load_frag(&qa[0][32], 72);   // K(d) = 32..63

  // each wave owns a 256-wide stripe of kv columns: 16 tiles of N=16
  const bf16* khp = kh + (size_t)bh * SEQ * HD;   // [kv][64] == BT (N-major)
  for (int t = 0; t < 16; ++t) {
    const int kv0 = (wave * 16 + t) * 16;
    const v16bf b0 = load_frag(khp + (size_t)kv0 * HD,      HD);
    const v16bf b1 = load_frag(khp + (size_t)kv0 * HD + 32, HD);
    v8f s = {};
    s = wmma_bf16(a0, b0, s);
    s = wmma_bf16(a1, b1, s);
    const int col = kv0 + (lane & 15);
    const int rb  = (lane >> 4) << 3;
#pragma unroll
    for (int r = 0; r < 8; ++r)
      sc[rb + r][col] = s[r] * 0.125f;      // 1/sqrt(64)
  }
  __syncthreads();

  // softmax: 16 threads per row, 128 columns each; shfl reductions width 16
  const int row = tid >> 4;
  const int cs  = (tid & 15) * 128;
  const unsigned char* mrow = mask + (((size_t)b * SEQ) + q0 + row) * SEQ + cs;

  float mx = -3.0e38f;
  for (int c = 0; c < 128; ++c) {
    float v = sc[row][cs + c];
    if (mrow[c]) v = -1.0e9f;               // masked => -1e9 (reference semantics)
    sc[row][cs + c] = v;
    mx = fmaxf(mx, v);
  }
#pragma unroll
  for (int o = 1; o < 16; o <<= 1) mx = fmaxf(mx, __shfl_xor(mx, o, 16));

  float sum = 0.f;
  for (int c = 0; c < 128; ++c) {
    float e = __expf(sc[row][cs + c] - mx);
    sc[row][cs + c] = e;
    sum += e;
  }
#pragma unroll
  for (int o = 1; o < 16; o <<= 1) sum += __shfl_xor(sum, o, 16);
  const float inv = 1.0f / sum;

  float* arow = attn + (((size_t)bh * SEQ) + q0 + row) * SEQ + cs;
  for (int c = 0; c < 128; ++c) arow[c] = sc[row][cs + c] * inv;
}

// ---- context = attn @ V  -> ctx bf16 [B,S,H*64] --------------------------
// A = attn f32 (converted to bf16 while staging), B = vhT [B,H,64,S] (N-major).

__global__ __launch_bounds__(256) void context_gemm(
    const float* __restrict__ attn, const bf16* __restrict__ vhT,
    bf16* __restrict__ ctx) {
  __shared__ bf16 As[128][40];
  __shared__ bf16 Bs[64][40];

  const int bh = blockIdx.y;
  const int m0 = blockIdx.x * 128;
  const int b  = bh >> 4, h = bh & 15;
  const int tid  = threadIdx.x;
  const int lane = tid & 31;
  const int wave = tid >> 5;     // 8 waves: 4 (M) x 2 (N)
  const int wm = wave >> 1, wn = wave & 1;

  const float* Ap = attn + (size_t)bh * SEQ * SEQ;
  const bf16*  Bp = vhT  + (size_t)bh * HD * SEQ;

  v8f acc[2][2] = {};

  for (int kt = 0; kt < SEQ; kt += 32) {
    {  // stage A: 128x32 f32 -> bf16
      const int row = tid >> 1, cb = (tid & 1) * 16;
      const float* src = Ap + (size_t)(m0 + row) * SEQ + kt + cb;
#pragma unroll
      for (int i = 0; i < 16; ++i) As[row][cb + i] = (bf16)src[i];
    }
    {  // stage B: 64x32 bf16 (already N-major)
      const int n = tid >> 2, cb = (tid & 3) * 8;
      *(v8bf*)&Bs[n][cb] = *(const v8bf*)(Bp + (size_t)n * SEQ + kt + cb);
    }
    __syncthreads();

    v16bf af[2], bfg[2];
#pragma unroll
    for (int mi = 0; mi < 2; ++mi) af[mi]  = load_frag(&As[wm * 32 + mi * 16][0], 40);
#pragma unroll
    for (int nj = 0; nj < 2; ++nj) bfg[nj] = load_frag(&Bs[wn * 32 + nj * 16][0], 40);
#pragma unroll
    for (int mi = 0; mi < 2; ++mi)
#pragma unroll
      for (int nj = 0; nj < 2; ++nj)
        acc[mi][nj] = wmma_bf16(af[mi], bfg[nj], acc[mi][nj]);
    __syncthreads();
  }

#pragma unroll
  for (int mi = 0; mi < 2; ++mi) {
#pragma unroll
    for (int nj = 0; nj < 2; ++nj) {
      const int gn = wn * 32 + nj * 16 + (lane & 15);        // d: 0..63
#pragma unroll
      for (int r = 0; r < 8; ++r) {
        const int gm = m0 + wm * 32 + mi * 16 + r + ((lane >> 4) << 3);  // s
        ctx[((size_t)(b * SEQ + gm)) * DMODEL + h * HD + gn] = (bf16)acc[mi][nj][r];
      }
    }
  }
}

// ---- LayerNorm over D=1024 (eps 1e-5, population variance) ---------------

__global__ __launch_bounds__(256) void layernorm_rows(
    const float* __restrict__ x, const float* __restrict__ gamma,
    const float* __restrict__ beta, float* __restrict__ out) {
  __shared__ float s1[8], s2[8], mu_s, rs_s;
  const int row = blockIdx.x, tid = threadIdx.x;
  const float* xr = x + (size_t)row * DMODEL;

  float v[4], sum = 0.f, sq = 0.f;
#pragma unroll
  for (int i = 0; i < 4; ++i) {
    v[i] = xr[tid + 256 * i];
    sum += v[i];
    sq  += v[i] * v[i];
  }
#pragma unroll
  for (int o = 16; o >= 1; o >>= 1) {
    sum += __shfl_xor(sum, o, 32);
    sq  += __shfl_xor(sq,  o, 32);
  }
  if ((tid & 31) == 0) { s1[tid >> 5] = sum; s2[tid >> 5] = sq; }
  __syncthreads();
  if (tid == 0) {
    float S = 0.f, Q = 0.f;
#pragma unroll
    for (int i = 0; i < 8; ++i) { S += s1[i]; Q += s2[i]; }
    const float mu = S * (1.0f / DMODEL);
    mu_s = mu;
    rs_s = rsqrtf(Q * (1.0f / DMODEL) - mu * mu + 1e-5f);
  }
  __syncthreads();
  const float mu = mu_s, rs = rs_s;
#pragma unroll
  for (int i = 0; i < 4; ++i) {
    const int c = tid + 256 * i;
    out[(size_t)row * DMODEL + c] = (v[i] - mu) * rs * gamma[c] + beta[c];
  }
}

// ---------------------------------------------------------------------------

extern "C" void kernel_launch(void* const* d_in, const int* in_sizes, int n_in,
                              void* d_out, int out_size, void* d_ws, size_t ws_size,
                              hipStream_t stream) {
  const float* q    = (const float*)d_in[0];
  const float* k    = (const float*)d_in[1];
  const float* v    = (const float*)d_in[2];
  const unsigned char* mask = (const unsigned char*)d_in[3];   // bool array, 1 byte/elt
  const float* wq = (const float*)d_in[4];  const float* bq = (const float*)d_in[5];
  const float* wk = (const float*)d_in[6];  const float* bk = (const float*)d_in[7];
  const float* wv = (const float*)d_in[8];  const float* bv = (const float*)d_in[9];
  const float* wo = (const float*)d_in[10]; const float* bo = (const float*)d_in[11];
  const float* gamma = (const float*)d_in[12];
  const float* beta  = (const float*)d_in[13];

  float* outLN   = (float*)d_out;                              // [B,S,D] f32
  float* attnOut = outLN + (size_t)BATCH * SEQ * DMODEL;       // [B,H,S,S] f32

  // workspace carve (~80 MB)
  char* w = (char*)d_ws;
  auto carve = [&](size_t bytes) {
    char* p = w;
    w += (bytes + 255) & ~(size_t)255;
    return p;
  };
  const size_t MKb = (size_t)MROWS * DMODEL;        // 4M elements
  const size_t WKb = (size_t)DMODEL * DMODEL;       // 1M elements
  bf16* qbf = (bf16*)carve(MKb * 2);
  bf16* kbf = (bf16*)carve(MKb * 2);
  bf16* vbf = (bf16*)carve(MKb * 2);
  bf16* wqT = (bf16*)carve(WKb * 2);
  bf16* wkT = (bf16*)carve(WKb * 2);
  bf16* wvT = (bf16*)carve(WKb * 2);
  bf16* woT = (bf16*)carve(WKb * 2);
  bf16* qh  = (bf16*)carve(MKb * 2);   // [B,H,S,64]
  bf16* kh  = (bf16*)carve(MKb * 2);   // [B,H,S,64]
  bf16* vhT = (bf16*)carve(MKb * 2);   // [B,H,64,S]
  bf16* ctx = (bf16*)carve(MKb * 2);   // [B,S,H*64]
  float* outPre = (float*)carve(MKb * 4);

  // 1) dtype prep
  cvt_f32_bf16<<<(int)(MKb / 1024), 256, 0, stream>>>(q, qbf, (int)MKb);
  cvt_f32_bf16<<<(int)(MKb / 1024), 256, 0, stream>>>(k, kbf, (int)MKb);
  cvt_f32_bf16<<<(int)(MKb / 1024), 256, 0, stream>>>(v, vbf, (int)MKb);
  tcvt_f32_bf16<<<(int)(WKb / 256), 256, 0, stream>>>(wq, wqT, DMODEL, DMODEL);
  tcvt_f32_bf16<<<(int)(WKb / 256), 256, 0, stream>>>(wk, wkT, DMODEL, DMODEL);
  tcvt_f32_bf16<<<(int)(WKb / 256), 256, 0, stream>>>(wv, wvT, DMODEL, DMODEL);
  tcvt_f32_bf16<<<(int)(WKb / 256), 256, 0, stream>>>(wo, woT, DMODEL, DMODEL);

  // 2) Q/K/V projections (WMMA), scatter to head layouts
  dim3 gg(DMODEL / 128, MROWS / 128);   // (8, 32)
  gemm128<0><<<gg, 256, 0, stream>>>(qbf, wqT, bq, nullptr, nullptr, qh,
                                     MROWS, DMODEL, DMODEL);
  gemm128<0><<<gg, 256, 0, stream>>>(kbf, wkT, bk, nullptr, nullptr, kh,
                                     MROWS, DMODEL, DMODEL);
  gemm128<1><<<gg, 256, 0, stream>>>(vbf, wvT, bv, nullptr, nullptr, vhT,
                                     MROWS, DMODEL, DMODEL);

  // 3) scores + mask + softmax entirely in LDS; write normalized attn once
  attn_scores_softmax<<<dim3(SEQ / 16, BATCH * NH), 256, 0, stream>>>(
      qh, kh, mask, attnOut);

  // 4) context = attn @ V (WMMA)
  context_gemm<<<dim3(SEQ / 128, BATCH * NH), 256, 0, stream>>>(attnOut, vhT, ctx);

  // 5) output projection + bias + residual (WMMA)
  gemm128<2><<<gg, 256, 0, stream>>>(ctx, woT, bo, q, outPre, nullptr,
                                     MROWS, DMODEL, DMODEL);

  // 6) LayerNorm
  layernorm_rows<<<MROWS, 256, 0, stream>>>(outPre, gamma, beta, outLN);
}
